// MultiHeadAttention_38276748542791
// MI455X (gfx1250) — matrix operand order repaired and checked by reading, not compiled, not run on hardware
//
#include <hip/hip_runtime.h>

// Problem sizes (fixed by the reference)
#define B_  2
#define S_  2048
#define D_  1024
#define H_  16
#define DK_ 64

typedef __attribute__((ext_vector_type(16))) _Float16 v16h;
typedef __attribute__((ext_vector_type(8)))  _Float16 v8h;
typedef __attribute__((ext_vector_type(8)))  float    v8f;

// D = A(16x32 f16) * B(32x16 f16) + C(16x16 f32)
#define WMMA_F16(a, b, c) \
  __builtin_amdgcn_wmma_f32_16x16x32_f16(false, (a), false, (b), (short)0, (c), false, false)

// ---------------------------------------------------------------------------
// Fragment loaders (wave32). lane = 32 lanes; ln = lane&15, lh = lane>>4.
//
// A fragment 16x32 f16: lane ln = row, lane-half lh selects K-group:
//   halves 0..15 <- K = 16*lh + 0..15 (same order as B)
// B fragment 32x16 f16: lane ln = column, halves 0..15 <- K = 16*lh + 0..15
//   (column data must be contiguous in memory: ptr = base + ln*ldb)
// C/D fragment 16x16 f32: c[j] = C[8*lh + j][ln]
// ---------------------------------------------------------------------------

__device__ __forceinline__ v16h load_a_f16(const _Float16* __restrict__ p, int lda, int ln, int lh) {
  const _Float16* r = p + (size_t)ln * lda + lh * 16;
  v8h lo = *(const v8h*)(r);
  v8h hi = *(const v8h*)(r + 8);
  v16h a;
#pragma unroll
  for (int i = 0; i < 8; ++i) { a[i] = lo[i]; a[8 + i] = hi[i]; }
  return a;
}

__device__ __forceinline__ v16h load_a_f32(const float* __restrict__ p, int lda, int ln, int lh) {
  const float* r = p + (size_t)ln * lda + lh * 16;
  float4 l0 = *(const float4*)(r);
  float4 l1 = *(const float4*)(r + 4);
  float4 h0 = *(const float4*)(r + 8);
  float4 h1 = *(const float4*)(r + 12);
  v16h a;
  a[0]  = (_Float16)l0.x; a[1]  = (_Float16)l0.y; a[2]  = (_Float16)l0.z; a[3]  = (_Float16)l0.w;
  a[4]  = (_Float16)l1.x; a[5]  = (_Float16)l1.y; a[6]  = (_Float16)l1.z; a[7]  = (_Float16)l1.w;
  a[8]  = (_Float16)h0.x; a[9]  = (_Float16)h0.y; a[10] = (_Float16)h0.z; a[11] = (_Float16)h0.w;
  a[12] = (_Float16)h1.x; a[13] = (_Float16)h1.y; a[14] = (_Float16)h1.z; a[15] = (_Float16)h1.w;
  return a;
}

__device__ __forceinline__ v16h load_b_f16(const _Float16* __restrict__ p, int ldb, int ln, int lh) {
  return *(const v16h*)(p + (size_t)ln * ldb + lh * 16);
}

// 16 consecutive f32 -> 16 f16 (for staging weight tiles into LDS)
__device__ __forceinline__ v16h cvt16_f32(const float* __restrict__ src) {
  float4 t0 = *(const float4*)(src);
  float4 t1 = *(const float4*)(src + 4);
  float4 t2 = *(const float4*)(src + 8);
  float4 t3 = *(const float4*)(src + 12);
  v16h h;
  h[0]  = (_Float16)t0.x; h[1]  = (_Float16)t0.y; h[2]  = (_Float16)t0.z; h[3]  = (_Float16)t0.w;
  h[4]  = (_Float16)t1.x; h[5]  = (_Float16)t1.y; h[6]  = (_Float16)t1.z; h[7]  = (_Float16)t1.w;
  h[8]  = (_Float16)t2.x; h[9]  = (_Float16)t2.y; h[10] = (_Float16)t2.z; h[11] = (_Float16)t2.w;
  h[12] = (_Float16)t3.x; h[13] = (_Float16)t3.y; h[14] = (_Float16)t3.z; h[15] = (_Float16)t3.w;
  return h;
}

// ---------------------------------------------------------------------------
// Kernel 1: QKV projections.  x @ W^T + b  (W stored [out,in] -> B columns
// are W rows, contiguous).  z = 0/1/2 selects Q/K/V.  Q,K written f16
// row-major [b*S+s][D]; V written transposed vT[b][h][d][s] for attn@V.
// Block = 128 thr (4 waves, 2x2 of 32x32 tiles over a 64x64 block tile).
// W tile (64 cols x 32 K) staged f32->f16 in LDS, shared by all 4 waves.
// ---------------------------------------------------------------------------
__global__ void proj_qkv_kernel(const float* __restrict__ Qm, const float* __restrict__ Km,
                                const float* __restrict__ Vm,
                                const float* __restrict__ WQ, const float* __restrict__ WK,
                                const float* __restrict__ WV,
                                const float* __restrict__ bQ, const float* __restrict__ bK,
                                const float* __restrict__ bV,
                                _Float16* __restrict__ qh, _Float16* __restrict__ kh,
                                _Float16* __restrict__ vT) {
  __shared__ _Float16 Bs[64 * 32] __attribute__((aligned(32)));
  const int tid = threadIdx.x;
  const int lane = tid & 31, w = tid >> 5;
  const int ln = lane & 15, lh = lane >> 4;
  const int m0  = blockIdx.y * 64 + (w >> 1) * 32;
  const int n0g = blockIdx.x * 64;
  const int nb  = (w & 1) * 32;
  const int n0  = n0g + nb;

  const float* X; const float* W; const float* bias;
  if (blockIdx.z == 0)      { X = Qm; W = WQ; bias = bQ; }
  else if (blockIdx.z == 1) { X = Km; W = WK; bias = bK; }
  else                      { X = Vm; W = WV; bias = bV; }

  const int nl    = tid >> 1;        // 0..63: staged column
  const int khalf = (tid & 1) * 16;  // 0 or 16: staged K half

  v8f c00 = {}, c01 = {}, c10 = {}, c11 = {};
  for (int kk = 0; kk < D_; kk += 32) {
    // stage 64x32 weight tile into LDS, column-major Bs[n*32 + k]
    *(v16h*)&Bs[nl * 32 + khalf] = cvt16_f32(W + (size_t)(n0g + nl) * D_ + kk + khalf);
    __syncthreads();

    v16h a0 = load_a_f32(X + (size_t)m0 * D_ + kk, D_, ln, lh);
    v16h a1 = load_a_f32(X + (size_t)(m0 + 16) * D_ + kk, D_, ln, lh);
    v16h b0 = *(const v16h*)&Bs[(nb + ln) * 32 + 16 * lh];
    v16h b1 = *(const v16h*)&Bs[(nb + 16 + ln) * 32 + 16 * lh];
    c00 = WMMA_F16(a0, b0, c00);
    c01 = WMMA_F16(a0, b1, c01);
    c10 = WMMA_F16(a1, b0, c10);
    c11 = WMMA_F16(a1, b1, c11);
    __syncthreads();
  }

  const float bn0 = bias[n0 + ln];
  const float bn1 = bias[n0 + 16 + ln];

  if (blockIdx.z < 2) {
    _Float16* dst = (blockIdx.z == 0) ? qh : kh;
#pragma unroll
    for (int j = 0; j < 8; ++j) {
      dst[(size_t)(m0 + 8 * lh + j) * D_ + n0 + ln]           = (_Float16)(c00[j] + bn0);
      dst[(size_t)(m0 + 8 * lh + j) * D_ + n0 + 16 + ln]      = (_Float16)(c01[j] + bn1);
      dst[(size_t)(m0 + 16 + 8 * lh + j) * D_ + n0 + ln]      = (_Float16)(c10[j] + bn0);
      dst[(size_t)(m0 + 16 + 8 * lh + j) * D_ + n0 + 16 + ln] = (_Float16)(c11[j] + bn1);
    }
  } else {
    // vT[((b*H + h)*DK + d)*S + s]
    const int bb = m0 >> 11;                 // m = b*S + s, S = 2048
    const int s0 = (m0 & (S_ - 1)) + 8 * lh;
    const int na = n0 + ln, nbig = n0 + 16 + ln;
    _Float16* pa = vT + ((size_t)(bb * H_ + (na >> 6)) * DK_ + (na & 63)) * S_ + s0;
    _Float16* pb = vT + ((size_t)(bb * H_ + (nbig >> 6)) * DK_ + (nbig & 63)) * S_ + s0;
#pragma unroll
    for (int j = 0; j < 8; ++j) {
      pa[j]      = (_Float16)(c00[j] + bn0);
      pb[j]      = (_Float16)(c01[j] + bn1);
      pa[16 + j] = (_Float16)(c10[j] + bn0);
      pb[16 + j] = (_Float16)(c11[j] + bn1);
    }
  }
}

// ---------------------------------------------------------------------------
// Kernel 2: scores = (q @ k^T) / sqrt(DK), f32, written into the attn slab
// of d_out.  B columns = k rows (contiguous along head dim).
// ---------------------------------------------------------------------------
__global__ void scores_kernel(const _Float16* __restrict__ qh, const _Float16* __restrict__ kh,
                              float* __restrict__ attn) {
  const int tid = threadIdx.x, lane = tid & 31, w = tid >> 5;
  const int ln = lane & 15, lh = lane >> 4;
  const int bh = blockIdx.z, bb = bh >> 4, h = bh & 15;
  const int q0 = blockIdx.y * 64 + (w >> 1) * 32;
  const int k0 = blockIdx.x * 64 + (w & 1) * 32;
  const _Float16* qb = qh + ((size_t)bb * S_ + q0) * D_ + h * DK_;
  const _Float16* kb = kh + ((size_t)bb * S_ + k0) * D_ + h * DK_;

  v8f c00 = {}, c01 = {}, c10 = {}, c11 = {};
#pragma unroll
  for (int kk = 0; kk < DK_; kk += 32) {
    v16h a0 = load_a_f16(qb + kk, D_, ln, lh);
    v16h a1 = load_a_f16(qb + (size_t)16 * D_ + kk, D_, ln, lh);
    v16h b0 = load_b_f16(kb + kk, D_, ln, lh);
    v16h b1 = load_b_f16(kb + (size_t)16 * D_ + kk, D_, ln, lh);
    c00 = WMMA_F16(a0, b0, c00);
    c01 = WMMA_F16(a0, b1, c01);
    c10 = WMMA_F16(a1, b0, c10);
    c11 = WMMA_F16(a1, b1, c11);
  }

  const float sc = 0.125f;  // 1/sqrt(64)
  float* ob = attn + (size_t)bh * S_ * S_;
#pragma unroll
  for (int j = 0; j < 8; ++j) {
    ob[(size_t)(q0 + 8 * lh + j) * S_ + k0 + ln]           = c00[j] * sc;
    ob[(size_t)(q0 + 8 * lh + j) * S_ + k0 + 16 + ln]      = c01[j] * sc;
    ob[(size_t)(q0 + 16 + 8 * lh + j) * S_ + k0 + ln]      = c10[j] * sc;
    ob[(size_t)(q0 + 16 + 8 * lh + j) * S_ + k0 + 16 + ln] = c11[j] * sc;
  }
}

// ---------------------------------------------------------------------------
// Kernel 3: in-place row softmax over 2048 f32.  One wave32 per row; the row
// (64 floats/lane) stays in registers; wave reductions via shfl_xor.
// ---------------------------------------------------------------------------
__global__ void softmax_kernel(float* __restrict__ attn) {
  const int w = threadIdx.x >> 5, lane = threadIdx.x & 31;
  const size_t row = (size_t)blockIdx.x * 8 + w;
  float4* p = (float4*)(attn + row * S_);

  float4 v[16];
  float mx = -3.402823466e38f;
#pragma unroll
  for (int i = 0; i < 16; ++i) {
    v[i] = p[lane + i * 32];
    mx = fmaxf(mx, fmaxf(fmaxf(v[i].x, v[i].y), fmaxf(v[i].z, v[i].w)));
  }
#pragma unroll
  for (int off = 16; off >= 1; off >>= 1) mx = fmaxf(mx, __shfl_xor(mx, off, 32));

  float sum = 0.f;
#pragma unroll
  for (int i = 0; i < 16; ++i) {
    v[i].x = __expf(v[i].x - mx); v[i].y = __expf(v[i].y - mx);
    v[i].z = __expf(v[i].z - mx); v[i].w = __expf(v[i].w - mx);
    sum += v[i].x + v[i].y + v[i].z + v[i].w;
  }
#pragma unroll
  for (int off = 16; off >= 1; off >>= 1) sum += __shfl_xor(sum, off, 32);

  const float inv = 1.f / sum;
#pragma unroll
  for (int i = 0; i < 16; ++i) {
    v[i].x *= inv; v[i].y *= inv; v[i].z *= inv; v[i].w *= inv;
    p[lane + i * 32] = v[i];
  }
}

// ---------------------------------------------------------------------------
// Kernel 4: ctx = attn @ v.  A = attn rows (f32 -> f16 on load), B columns
// come from vT[b][h][d][s] (contiguous along s).  Output ctx f16 in natural
// [b*S+s][h*DK+d] layout for the final projection.
// ---------------------------------------------------------------------------
__global__ void attnv_kernel(const float* __restrict__ attn, const _Float16* __restrict__ vT,
                             _Float16* __restrict__ ctx) {
  const int tid = threadIdx.x, lane = tid & 31, w = tid >> 5;
  const int ln = lane & 15, lh = lane >> 4;
  const int bh = blockIdx.z, bb = bh >> 4, h = bh & 15;
  const int q0 = blockIdx.y * 64 + (w >> 1) * 32;
  const int d0 = (w & 1) * 32;
  const float* ab = attn + (size_t)bh * S_ * S_ + (size_t)q0 * S_;
  const _Float16* vb = vT + ((size_t)bh * DK_ + d0) * S_;

  v8f c00 = {}, c01 = {}, c10 = {}, c11 = {};
  for (int kk = 0; kk < S_; kk += 32) {
    __builtin_prefetch(ab + kk + 128, 0, 1);   // global_prefetch_b8 ahead of the stream
    v16h a0 = load_a_f32(ab + kk, S_, ln, lh);
    v16h a1 = load_a_f32(ab + (size_t)16 * S_ + kk, S_, ln, lh);
    v16h b0 = load_b_f16(vb + kk, S_, ln, lh);
    v16h b1 = load_b_f16(vb + (size_t)16 * S_ + kk, S_, ln, lh);
    c00 = WMMA_F16(a0, b0, c00);
    c01 = WMMA_F16(a0, b1, c01);
    c10 = WMMA_F16(a1, b0, c10);
    c11 = WMMA_F16(a1, b1, c11);
  }

  _Float16* cb = ctx + ((size_t)bb * S_ + q0 + 8 * lh) * D_ + h * DK_ + d0 + ln;
#pragma unroll
  for (int j = 0; j < 8; ++j) {
    cb[(size_t)j * D_]             = (_Float16)c00[j];
    cb[(size_t)j * D_ + 16]        = (_Float16)c01[j];
    cb[((size_t)j + 16) * D_]      = (_Float16)c10[j];
    cb[((size_t)j + 16) * D_ + 16] = (_Float16)c11[j];
  }
}

// ---------------------------------------------------------------------------
// Kernel 5: output = ctx @ W_O^T + b_O, f32 into d_out[0 .. B*S*D).
// Same LDS-staged weight-tile scheme as the projections.
// ---------------------------------------------------------------------------
__global__ void oproj_kernel(const _Float16* __restrict__ ctx, const float* __restrict__ WO,
                             const float* __restrict__ bO, float* __restrict__ out) {
  __shared__ _Float16 Bs[64 * 32] __attribute__((aligned(32)));
  const int tid = threadIdx.x, lane = tid & 31, w = tid >> 5;
  const int ln = lane & 15, lh = lane >> 4;
  const int m0  = blockIdx.y * 64 + (w >> 1) * 32;
  const int n0g = blockIdx.x * 64;
  const int nb  = (w & 1) * 32;
  const int n0  = n0g + nb;
  const int nl    = tid >> 1;
  const int khalf = (tid & 1) * 16;

  v8f c00 = {}, c01 = {}, c10 = {}, c11 = {};
  for (int kk = 0; kk < D_; kk += 32) {
    *(v16h*)&Bs[nl * 32 + khalf] = cvt16_f32(WO + (size_t)(n0g + nl) * D_ + kk + khalf);
    __syncthreads();

    v16h a0 = load_a_f16(ctx + (size_t)m0 * D_ + kk, D_, ln, lh);
    v16h a1 = load_a_f16(ctx + (size_t)(m0 + 16) * D_ + kk, D_, ln, lh);
    v16h b0 = *(const v16h*)&Bs[(nb + ln) * 32 + 16 * lh];
    v16h b1 = *(const v16h*)&Bs[(nb + 16 + ln) * 32 + 16 * lh];
    c00 = WMMA_F16(a0, b0, c00);
    c01 = WMMA_F16(a0, b1, c01);
    c10 = WMMA_F16(a1, b0, c10);
    c11 = WMMA_F16(a1, b1, c11);
    __syncthreads();
  }

  const float bn0 = bO[n0 + ln], bn1 = bO[n0 + 16 + ln];
#pragma unroll
  for (int j = 0; j < 8; ++j) {
    out[(size_t)(m0 + 8 * lh + j) * D_ + n0 + ln]           = c00[j] + bn0;
    out[(size_t)(m0 + 8 * lh + j) * D_ + n0 + 16 + ln]      = c01[j] + bn1;
    out[(size_t)(m0 + 16 + 8 * lh + j) * D_ + n0 + ln]      = c10[j] + bn0;
    out[(size_t)(m0 + 16 + 8 * lh + j) * D_ + n0 + 16 + ln] = c11[j] + bn1;
  }
}

// ---------------------------------------------------------------------------
extern "C" void kernel_launch(void* const* d_in, const int* in_sizes, int n_in,
                              void* d_out, int out_size, void* d_ws, size_t ws_size,
                              hipStream_t stream) {
  (void)in_sizes; (void)n_in; (void)out_size; (void)ws_size;
  const float* Q  = (const float*)d_in[0];
  const float* K  = (const float*)d_in[1];
  const float* V  = (const float*)d_in[2];
  const float* WQ = (const float*)d_in[3];
  const float* bQ = (const float*)d_in[4];
  const float* WK = (const float*)d_in[5];
  const float* bK = (const float*)d_in[6];
  const float* WV = (const float*)d_in[7];
  const float* bV = (const float*)d_in[8];
  const float* WO = (const float*)d_in[9];
  const float* bO = (const float*)d_in[10];

  float* out  = (float*)d_out;                        // (B,S,D) f32
  float* attn = out + (size_t)B_ * S_ * D_;           // (B,H,S,S) f32

  const size_t NE = (size_t)B_ * S_ * D_;             // 4,194,304 elements
  _Float16* qh  = (_Float16*)d_ws;                    //  8 MB
  _Float16* kh  = qh + NE;                            //  8 MB
  _Float16* vT  = kh + NE;                            //  8 MB  (total 24 MB ws)
  _Float16* ctx = qh;                                 // qh dead after scores -> reuse

  dim3 blk(128);
  proj_qkv_kernel<<<dim3(D_ / 64, (B_ * S_) / 64, 3), blk, 0, stream>>>(
      Q, K, V, WQ, WK, WV, bQ, bK, bV, qh, kh, vT);
  scores_kernel<<<dim3(S_ / 64, S_ / 64, B_ * H_), blk, 0, stream>>>(qh, kh, attn);
  softmax_kernel<<<dim3((B_ * H_ * S_) / 8), dim3(256), 0, stream>>>(attn);
  attnv_kernel<<<dim3(1, S_ / 64, B_ * H_), blk, 0, stream>>>(attn, vT, ctx);
  oproj_kernel<<<dim3(D_ / 64, (B_ * S_) / 64), blk, 0, stream>>>(ctx, WO, bO, out);
}
